// AsymmetricCausalSelfAttention_25683904430762
// MI455X (gfx1250) — matrix operand order repaired and checked by reading, not compiled, not run on hardware
//
#include <hip/hip_runtime.h>

typedef unsigned short u16;
typedef __attribute__((ext_vector_type(16))) __bf16 v16bf;
typedef __attribute__((ext_vector_type(8)))  float  v8f;
typedef __attribute__((ext_vector_type(4)))  unsigned int u32x4;

// Model dims
#define BB 2
#define TT 2048
#define CC 1024
#define HH 16
#define DD 64
#define BHTD (BB*HH*TT*DD)   // 4194304 elements per q/k/v tensor

union Frag { v16bf v; u32x4 q[2]; u16 s[16]; };

__device__ inline v8f wmma_bf16(const Frag& a, const Frag& b, v8f c) {
  return __builtin_amdgcn_wmma_f32_16x16x32_bf16(false, a.v, false, b.v,
                                                 (short)0, c, false, false);
}

__device__ inline u16 f32_to_bf16(float f) {
  unsigned u = __builtin_bit_cast(unsigned, f);
  u += 0x7FFFu + ((u >> 16) & 1u);           // round-to-nearest-even
  return (u16)(u >> 16);
}

__device__ inline v8f zero8() { v8f z; for (int i = 0; i < 8; ++i) z[i] = 0.f; return z; }

// ---------------------------------------------------------------- conversion
__global__ void cvt_f32_bf16(const float* __restrict__ src, u16* __restrict__ dst, int n) {
  int i = blockIdx.x * blockDim.x + threadIdx.x;
  if (i < n) dst[i] = f32_to_bf16(src[i]);
}

// ---------------------------------------------------------------- GEMM 64x64 tile
// MODE 0: C = A@B + bias, scatter bf16 into qkv [3][B,H,T,D]
// MODE 1: C = A@B + bias, f32 row-major out
// Double-buffered B staging -> single barrier per K-step.
template<int MODE>
__global__ __launch_bounds__(128)
void gemm_bf16_wmma(const u16* __restrict__ A, const u16* __restrict__ Bm,
                    const float* __restrict__ bias,
                    u16* __restrict__ outb, float* __restrict__ outf,
                    int Kdim, int Ndim) {
  __shared__ u16 ldsB[2][4 * 512];         // ping-pong; 4 B-fragments each
  const int tid  = threadIdx.x;
  const int wave = tid >> 5, lane = tid & 31;
  const int ln = lane & 15, lh = lane >> 4;
  const int rowBase = blockIdx.y * 64 + wave * 16;
  const int colBase = blockIdx.x * 64;

  v8f acc[4];
  #pragma unroll
  for (int c = 0; c < 4; ++c) acc[c] = zero8();

  const u16* arow = A + (size_t)(rowBase + ln) * Kdim;
  const int bk = tid >> 2, bq = tid & 3;   // staging: k-row 0..31, n-quarter 0..3
  int ibuf = 0;

  for (int kc = 0; kc < Kdim; kc += 32, ibuf ^= 1) {
    { // stage B block [32 x 64] -> fragment-swizzled LDS (ping-pong buffer)
      const u16* bp = Bm + (size_t)(kc + bk) * Ndim + colBase + bq * 16;
      Frag t; t.q[0] = *(const u32x4*)bp; t.q[1] = *(const u32x4*)(bp + 8);
      const int lbase = 16 * (bk >> 4);    // lanes 0-15 hold K 0-15, 16-31 hold 16-31
      const int e     = bk & 15;
      u16* dst = &ldsB[ibuf][bq * 512 + e];
      #pragma unroll
      for (int i = 0; i < 16; ++i) dst[(lbase + i) * 16] = t.s[i];
    }
    __syncthreads();                       // waits dscnt then barrier (see disasm)

    if (kc + 32 < Kdim)                    // prefetch next A chunk (L2-resident)
      __builtin_prefetch(arow + kc + 32, 0, 1);

    Frag af;                               // A-frag direct from global, two 16B chunks
    af.q[0] = *(const u32x4*)(arow + kc + 16 * lh);
    af.q[1] = *(const u32x4*)(arow + kc + 16 * lh + 8);

    #pragma unroll
    for (int c = 0; c < 4; ++c) {
      Frag bf;
      bf.q[0] = *(const u32x4*)&ldsB[ibuf][c * 512 + lane * 16];
      bf.q[1] = *(const u32x4*)&ldsB[ibuf][c * 512 + lane * 16 + 8];
      acc[c] = wmma_bf16(af, bf, acc[c]);
    }
  }

  #pragma unroll
  for (int c = 0; c < 4; ++c) {
    #pragma unroll
    for (int r = 0; r < 8; ++r) {
      const int row = rowBase + r + 8 * lh;     // C layout: M = r + 8*(lane/16)
      const int j   = colBase + c * 16 + ln;    //           N = lane&15 (+chunk)
      const float val = acc[c][r] + bias[j];
      if (MODE == 0) {
        const int part = j >> 10;               // 0=q 1=k 2=v
        const int rem  = j & 1023;
        const int h = rem >> 6, dd = rem & 63;
        const int b = row >> 11, t = row & 2047;
        const size_t idx = (size_t)part * BHTD +
                           ((((size_t)b * HH + h) * TT + t) * DD + dd);
        outb[idx] = f32_to_bf16(val);
      } else {
        outf[(size_t)row * Ndim + j] = val;
      }
    }
  }
}

// ---------------------------------------------------------------- attention
// grid: (T/64, B*H), block 128 (4 waves). Wave w owns 16 query rows.
// Key tile = 64. Softmax denominator computed by WMMA ones-trick (P @ 1),
// so only the row-max needs cross-lane shuffles.
__global__ __launch_bounds__(128)
void attn_bf16_wmma(const u16* __restrict__ qg, const u16* __restrict__ kg,
                    const u16* __restrict__ vg,
                    const float* __restrict__ head_temp,
                    const float* __restrict__ head_scale,
                    u16* __restrict__ yout) {
  __shared__ u16 ldsV[8 * 512];    // V tile [64x64] as 8 B-fragments (s2*4+ddchunk)
  __shared__ u16 ldsP[4 * 1024];   // per-wave P (16x64) as 2 A-fragments

  const int bh = blockIdx.y;                 // b*H + h
  const int b = bh >> 4, h = bh & 15;
  const int tid = threadIdx.x;
  const int wave = tid >> 5, lane = tid & 31;
  const int ln = lane & 15, lh = lane >> 4;

  const size_t base = (size_t)bh * TT * DD;
  const u16* Q = qg + base;
  const u16* K = kg + base;
  const u16* V = vg + base;

  const int qrow = blockIdx.x * 64 + wave * 16;

  // Q fragments for both K-steps over D=64 (resident all loop)
  Frag qf[2];
  {
    const u16* qp = Q + (size_t)(qrow + ln) * DD;
    #pragma unroll
    for (int s = 0; s < 2; ++s) {
      const int d0 = s * 32;
      qf[s].q[0] = *(const u32x4*)(qp + d0 + 16 * lh);
      qf[s].q[1] = *(const u32x4*)(qp + d0 + 16 * lh + 8);
    }
  }

  Frag onesf;                                // B = all ones -> D[m,n] = row-sum(A)
  #pragma unroll
  for (int i = 0; i < 16; ++i) onesf.s[i] = 0x3F80;  // bf16 1.0

  const float temp = head_temp[h];
  const float hs   = head_scale[h];

  float mrow[8];
  v8f accO[4], accL;
  #pragma unroll
  for (int r = 0; r < 8; ++r) mrow[r] = -1e30f;
  #pragma unroll
  for (int c = 0; c < 4; ++c) accO[c] = zero8();
  accL = zero8();

  for (int kt = 0; kt < TT; kt += 64) {
    __syncthreads();                         // protect ldsV from prior readers
    // stage V tile [64 x 64] -> 8 B-fragments; 2 units per thread
    #pragma unroll
    for (int u0 = 0; u0 < 2; ++u0) {
      const int u = tid + u0 * 128;
      const int kk = u >> 2, qn = u & 3;
      const u16* vp = V + (size_t)(kt + kk) * DD + qn * 16;
      Frag t; t.q[0] = *(const u32x4*)vp; t.q[1] = *(const u32x4*)(vp + 8);
      const int f    = (kk >> 5) * 4 + qn;
      const int kkr  = kk & 31;
      const int lbase = 16 * (kkr >> 4);
      const int e     = kkr & 15;
      u16* dst = &ldsV[f * 512 + e];
      #pragma unroll
      for (int i = 0; i < 16; ++i) dst[(lbase + i) * 16] = t.s[i];
    }
    __syncthreads();

    // scores S[16 x 64] = Q @ K^T  (K^T B-frags are contiguous 32B global reads)
    v8f sc[4];
    #pragma unroll
    for (int c = 0; c < 4; ++c) sc[c] = zero8();
    #pragma unroll
    for (int s2 = 0; s2 < 2; ++s2) {
      const int d0 = s2 * 32;
      #pragma unroll
      for (int c = 0; c < 4; ++c) {
        Frag kf;
        const u16* kp = K + (size_t)(kt + c * 16 + ln) * DD + d0 + lh * 16;
        kf.q[0] = *(const u32x4*)kp; kf.q[1] = *(const u32x4*)(kp + 8);
        sc[c] = wmma_bf16(qf[s2], kf, sc[c]);
      }
    }

    if (kt + 64 < TT) {                      // prefetch next key tile (L2-resident)
      __builtin_prefetch(K + (size_t)(kt + 64 + ln) * DD, 0, 1);
      __builtin_prefetch(V + (size_t)(kt + 64 + (tid >> 2)) * DD, 0, 1);
    }

    // online softmax: only row-max needs shuffles (sum comes from P@1 WMMA)
    float pv[4][8];
    #pragma unroll
    for (int r = 0; r < 8; ++r) {
      float x0 = sc[0][r] * temp, x1 = sc[1][r] * temp;
      float x2 = sc[2][r] * temp, x3 = sc[3][r] * temp;
      float mx = fmaxf(fmaxf(x0, x1), fmaxf(x2, x3));
      #pragma unroll
      for (int msk = 8; msk >= 1; msk >>= 1) mx = fmaxf(mx, __shfl_xor(mx, msk, 32));
      const float mnew  = fmaxf(mrow[r], mx);
      const float scale = __expf(mrow[r] - mnew);
      mrow[r] = mnew;
      pv[0][r] = __expf(x0 - mnew);
      pv[1][r] = __expf(x1 - mnew);
      pv[2][r] = __expf(x2 - mnew);
      pv[3][r] = __expf(x3 - mnew);
      #pragma unroll
      for (int c = 0; c < 4; ++c) accO[c][r] *= scale;
      accL[r] *= scale;
    }

    // transpose-stage P (16x64) -> 2 A-fragments (wave-private LDS region)
    {
      u16* pbase = &ldsP[wave * 1024];
      #pragma unroll
      for (int r = 0; r < 8; ++r) {
        const int M = r + 8 * lh;
        #pragma unroll
        for (int c = 0; c < 4; ++c) {
          const int kk  = ln + 16 * c;       // key col of sc[c] element
          const int f   = kk >> 5;
          const int kkr = kk & 31;
          pbase[f * 512 + (M + 16 * (kkr >> 4)) * 16 +
                (kkr & 15)] = f32_to_bf16(pv[c][r]);
        }
      }
      // same-wave LDS producer->consumer across lanes: wait writes landed
      asm volatile("s_wait_dscnt 0" ::: "memory");
    }
    Frag pf[2];
    #pragma unroll
    for (int s2 = 0; s2 < 2; ++s2) {
      pf[s2].q[0] = *(const u32x4*)&ldsP[wave * 1024 + s2 * 512 + lane * 16];
      pf[s2].q[1] = *(const u32x4*)&ldsP[wave * 1024 + s2 * 512 + lane * 16 + 8];
    }

    #pragma unroll
    for (int s2 = 0; s2 < 2; ++s2) {
      accL = wmma_bf16(pf[s2], onesf, accL);      // denominator: P @ 1
      #pragma unroll
      for (int c = 0; c < 4; ++c) {
        Frag vf;
        vf.q[0] = *(const u32x4*)&ldsV[(s2 * 4 + c) * 512 + lane * 16];
        vf.q[1] = *(const u32x4*)&ldsV[(s2 * 4 + c) * 512 + lane * 16 + 8];
        accO[c] = wmma_bf16(pf[s2], vf, accO[c]);
      }
    }
  }

  // epilogue: y[b,t,h*64+dd] = head_scale * O / l   (bf16 for proj GEMM)
  #pragma unroll
  for (int c = 0; c < 4; ++c) {
    #pragma unroll
    for (int r = 0; r < 8; ++r) {
      const int t  = qrow + r + 8 * lh;
      const int dd = c * 16 + ln;
      const float val = accO[c][r] * hs / accL[r];
      yout[((size_t)b * TT + t) * CC + h * DD + dd] = f32_to_bf16(val);
    }
  }
}

// ---------------------------------------------------------------- launch
extern "C" void kernel_launch(void* const* d_in, const int* in_sizes, int n_in,
                              void* d_out, int out_size, void* d_ws, size_t ws_size,
                              hipStream_t stream) {
  const float* x          = (const float*)d_in[0];
  const float* W_attn     = (const float*)d_in[1];
  const float* b_attn     = (const float*)d_in[2];
  const float* head_temp  = (const float*)d_in[3];
  const float* head_scale = (const float*)d_in[4];
  const float* W_proj     = (const float*)d_in[5];
  const float* b_proj     = (const float*)d_in[6];
  float* out = (float*)d_out;

  // workspace carve (all bf16 as u16), total 48 MB
  u16* ws   = (u16*)d_ws;
  u16* xb   = ws;                     // 4096*1024
  u16* Wab  = xb   + 4194304;         // 1024*3072
  u16* Wpb  = Wab  + 3145728;         // 1024*1024
  u16* qkvb = Wpb  + 1048576;         // 3 * 4194304
  u16* yb   = qkvb + 12582912;        // 4096*1024

  // 1) f32 -> bf16
  cvt_f32_bf16<<<(4194304 + 255) / 256, 256, 0, stream>>>(x, xb, 4194304);
  cvt_f32_bf16<<<(3145728 + 255) / 256, 256, 0, stream>>>(W_attn, Wab, 3145728);
  cvt_f32_bf16<<<(1048576 + 255) / 256, 256, 0, stream>>>(W_proj, Wpb, 1048576);

  // 2) QKV GEMM: [4096,1024] @ [1024,3072], scatter to q/k/v [B,H,T,D]
  gemm_bf16_wmma<0><<<dim3(3072 / 64, 4096 / 64), 128, 0, stream>>>(
      xb, Wab, b_attn, qkvb, nullptr, /*K=*/1024, /*N=*/3072);

  // 3) attention (no mask, per reference), y -> bf16 [B,T,C]
  attn_bf16_wmma<<<dim3(TT / 64, BB * HH), 128, 0, stream>>>(
      qkvb, qkvb + BHTD, qkvb + 2 * BHTD, head_temp, head_scale, yb);

  // 4) proj GEMM: [4096,1024] @ [1024,1024] + b_proj -> f32 out
  gemm_bf16_wmma<1><<<dim3(1024 / 64, 4096 / 64), 128, 0, stream>>>(
      yb, Wpb, b_proj, nullptr, out, /*K=*/1024, /*N=*/1024);

  (void)in_sizes; (void)n_in; (void)out_size; (void)ws_size;
}
